// CausalSelfAttention_18846316495118
// MI455X (gfx1250) — compile-verified, matrix operand order checked by reading, not run on hardware
//
#include <hip/hip_runtime.h>
#include <cstdint>

typedef __attribute__((ext_vector_type(16))) __bf16 v16bf;
typedef __attribute__((ext_vector_type(8)))  __bf16 v8bf;
typedef __attribute__((ext_vector_type(8)))  float  v8f;
typedef unsigned int u32;
typedef __attribute__((ext_vector_type(4))) u32 v4u;
typedef __attribute__((ext_vector_type(8))) int  v8i;
typedef __attribute__((ext_vector_type(4))) int  v4i;

// Problem constants (reference: B=2, T=2048, C=1024, 16 heads, 4 KV heads)
constexpr int Bn   = 2;
constexpr int T    = 2048;
constexpr int CD   = 1024;          // model dim
constexpr int H    = 16;
constexpr int HD   = 64;            // head dim
constexpr int KV   = 4;
constexpr int KVD  = KV * HD;       // 256
constexpr int MR   = Bn * T;        // 4096 rows

// ---------------------------------------------------------------------
// LDS byte offset of a shared-memory pointer: flat LDS addresses carry the
// LDS offset in addr[31:0] (ISA aperture mapping), which is what TDM wants.
__device__ __forceinline__ u32 lds_off(const void* p) {
  return (u32)(uintptr_t)p;
}

// Tensor Data Mover: DMA a 2D tile (tile_d0 contiguous elems x tile_d1 rows,
// row stride = stride_elems, element = 2 bytes) from global into LDS.
__device__ __forceinline__ void tdm_load_2d(u32 ldsoff, const void* gptr,
                                            u32 tile_d0, u32 tile_d1,
                                            u32 stride_elems) {
  const unsigned long long ga = (unsigned long long)(uintptr_t)gptr;
  v4u g0;
  g0[0] = 1u;                                              // count=1, user mode
  g0[1] = ldsoff;                                          // lds_addr
  g0[2] = (u32)ga;                                         // global_addr[31:0]
  g0[3] = ((u32)(ga >> 32) & 0x01FFFFFFu) | (2u << 30);    // addr[56:32] | type=2
  v8i g1;
  g1[0] = (int)(1u << 16);                                 // data_size = 2 bytes
  g1[1] = (int)((tile_d0 & 0xFFFFu) << 16);                // tensor_dim0[15:0]
  g1[2] = (int)((tile_d1 & 0xFFFFu) << 16);                // tensor_dim1[15:0]
  g1[3] = (int)((tile_d0 & 0xFFFFu) << 16);                // tile_dim0
  g1[4] = (int)(tile_d1 & 0xFFFFu);                        // tile_dim1 | tile_dim2=0
  g1[5] = (int)stride_elems;                               // tensor_dim0_stride[31:0]
  g1[6] = 0;
  g1[7] = 0;
  v4i z = {0, 0, 0, 0};
#if __clang_major__ >= 23
  v8i z8 = {0, 0, 0, 0, 0, 0, 0, 0};
  __builtin_amdgcn_tensor_load_to_lds(g0, g1, z, z, z8, 0);
#else
  __builtin_amdgcn_tensor_load_to_lds(g0, g1, z, z, 0);
#endif
}

// Build a 16x32 A/B fragment for this lane from two contiguous 16-byte runs:
// elements 0..7 = K(kb..kb+7), elements 8..15 = K(16+kb..16+kb+7).
__device__ __forceinline__ v16bf frag_from(const __bf16* base) {
  v8bf lo = *(const v8bf*)(base);
  v8bf hi = *(const v8bf*)(base + 16);
  return __builtin_shufflevector(lo, hi, 0, 1, 2, 3, 4, 5, 6, 7,
                                 8, 9, 10, 11, 12, 13, 14, 15);
}

// -------------------- fp32 -> bf16 conversion --------------------
__global__ void k_tobf16(const float* __restrict__ in, __bf16* __restrict__ out, int n) {
  int i = blockIdx.x * blockDim.x + threadIdx.x;
  int stride = gridDim.x * blockDim.x;
  for (; i < n; i += stride) out[i] = (__bf16)in[i];
}

// fp32 (K x N) -> bf16 transposed (N x K): makes GEMM B-fragments contiguous.
__global__ void k_tobf16T(const float* __restrict__ in, __bf16* __restrict__ out,
                          int K, int N) {
  int i = blockIdx.x * blockDim.x + threadIdx.x;
  const int total = K * N;
  const int stride = gridDim.x * blockDim.x;
  for (; i < total; i += stride) {
    const int k = i / N, n = i % N;
    out[(size_t)n * K + k] = (__bf16)in[i];
  }
}

// ------------- bf16 GEMM, TDM-staged, Wt is transposed (N x K) ---------
// One wave computes a 32x64 strip of C = A(MxK) * W(KxN). 8 waves / block.
// Per K-chunk: A tile 32x32 and Wt tile 64x32 staged by TDM (double
// buffered). All fragment loads hoisted ahead of the 8 WMMAs so ds traffic
// overlaps matrix execution.
__global__ void __launch_bounds__(256)
k_gemm(const __bf16* __restrict__ A, const __bf16* __restrict__ Wt,
       float* __restrict__ Cm, int Nd, int Kd) {
  // per wave, per buffer: A tile (32x32 = 1024) then Wt tile (64x32 = 2048)
  __shared__ __bf16 smem[8][2][3072];

  const int lane = threadIdx.x & 31;
  const int wsl  = threadIdx.x >> 5;
  const int wid  = blockIdx.x * 8 + wsl;
  const int tilesN = Nd >> 6;
  const int tm = wid / tilesN;       // 32-row strip
  const int tn = wid % tilesN;       // 64-col strip
  const int lrow = lane & 15;
  const int half = lane >> 4;
  const int kb   = half ? 8 : 0;

  const __bf16* Abase = A  + (size_t)(tm * 32) * Kd;
  const __bf16* Bbase = Wt + (size_t)(tn * 64) * Kd;

  tdm_load_2d(lds_off(&smem[wsl][0][0]),    Abase, 32, 32, (u32)Kd);
  tdm_load_2d(lds_off(&smem[wsl][0][1024]), Bbase, 32, 64, (u32)Kd);

  v8f acc[2][4] = {{{}, {}, {}, {}}, {{}, {}, {}, {}}};
  int buf = 0;
  for (int k0 = 0; k0 < Kd; k0 += 32) {
    if (k0 + 32 < Kd) {
      tdm_load_2d(lds_off(&smem[wsl][buf ^ 1][0]),    Abase + (k0 + 32), 32, 32, (u32)Kd);
      tdm_load_2d(lds_off(&smem[wsl][buf ^ 1][1024]), Bbase + (k0 + 32), 32, 64, (u32)Kd);
      __builtin_amdgcn_s_wait_tensorcnt(2);
    } else {
      __builtin_amdgcn_s_wait_tensorcnt(0);
    }
    asm volatile("" ::: "memory");

    const __bf16* As  = &smem[wsl][buf][0];       // 32 rows x 32 K
    const __bf16* Bts = As + 1024;                // 64 N-rows x 32 K

    // hoist all fragment loads before the WMMA burst
    const v16bf a0 = frag_from(As + lrow * 32 + kb);
    const v16bf a1 = frag_from(As + (16 + lrow) * 32 + kb);
    v16bf bfr[4];
#pragma unroll
    for (int t = 0; t < 4; ++t)
      bfr[t] = frag_from(Bts + (t * 16 + lrow) * 32 + kb);

#pragma unroll
    for (int t = 0; t < 4; ++t)
      acc[0][t] = __builtin_amdgcn_wmma_f32_16x16x32_bf16(false, a0, false, bfr[t],
                                                          (short)0, acc[0][t], false, false);
#pragma unroll
    for (int t = 0; t < 4; ++t)
      acc[1][t] = __builtin_amdgcn_wmma_f32_16x16x32_bf16(false, a1, false, bfr[t],
                                                          (short)0, acc[1][t], false, false);
    buf ^= 1;
  }

#pragma unroll
  for (int r = 0; r < 2; ++r)
#pragma unroll
    for (int t = 0; t < 4; ++t)
#pragma unroll
      for (int mp = 0; mp < 8; ++mp) {
        const int mrow = tm * 32 + r * 16 + mp + 8 * half;
        Cm[(size_t)mrow * Nd + tn * 64 + t * 16 + lrow] = acc[r][t][mp];
      }
}

// -------------------- RoPE on Q (fp32 in, bf16 out, (B,T,C) layout) ----
__global__ void k_rope_q(const float* __restrict__ qf, __bf16* __restrict__ qb) {
  const int n = MR * CD;
  int i = blockIdx.x * blockDim.x + threadIdx.x;
  const int stride = gridDim.x * blockDim.x;
  for (; i < n; i += stride) {
    const int c   = i % CD;
    const int row = i / CD;
    const int t   = row % T;
    const int d   = c & (HD - 1);
    const int j   = d & 31;
    const float ang = (float)t * __expf(-(float)(2 * j) * (9.210340371976184f / 64.f));
    float s, cs;
    __sincosf(ang, &s, &cs);
    const float x  = qf[i];
    const float xr = (d < 32) ? -qf[i + 32] : qf[i - 32];
    qb[i] = (__bf16)(x * cs + xr * s);
  }
}

// ---- RoPE on K: bf16 (B,T,kv*hd) for attention + fp32 cache (B,KV,T,hd) ----
__global__ void k_rope_k(const float* __restrict__ kf, __bf16* __restrict__ kbb,
                         float* __restrict__ kout) {
  const int n = MR * KVD;
  int i = blockIdx.x * blockDim.x + threadIdx.x;
  const int stride = gridDim.x * blockDim.x;
  for (; i < n; i += stride) {
    const int c   = i % KVD;
    const int row = i / KVD;
    const int b   = row / T;
    const int t   = row % T;
    const int g   = c >> 6;
    const int d   = c & 63;
    const int j   = d & 31;
    const float ang = (float)t * __expf(-(float)(2 * j) * (9.210340371976184f / 64.f));
    float s, cs;
    __sincosf(ang, &s, &cs);
    const float x   = kf[i];
    const float xr  = (d < 32) ? -kf[i + 32] : kf[i - 32];
    const float val = x * cs + xr * s;
    kbb[i] = (__bf16)val;
    kout[(((size_t)b * KV + g) * T + t) * HD + d] = val;
  }
}

// ---- V: bf16 d-major (b,g,d,t) for attention + fp32 cache (B,KV,T,hd) ----
__global__ void k_vcache(const float* __restrict__ vf, __bf16* __restrict__ vbt,
                         float* __restrict__ vout) {
  const int n = MR * KVD;
  int i = blockIdx.x * blockDim.x + threadIdx.x;
  const int stride = gridDim.x * blockDim.x;
  for (; i < n; i += stride) {
    const int c   = i % KVD;
    const int row = i / KVD;
    const int b   = row / T;
    const int t   = row % T;
    const int g   = c >> 6;
    const int d   = c & 63;
    const float val = vf[i];
    vbt[(((size_t)b * KV + g) * HD + d) * T + t] = (__bf16)val;   // d-major
    vout[(((size_t)b * KV + g) * T + t) * HD + d] = val;
  }
}

// -------------------- flash attention (causal, GQA) -------------------
// One wave owns a 16-query tile of one head. Keys processed 32 at a time.
// K (32x64, key-major) and V (64x32, d-major) tiles TDM-staged, double
// buffered. All fragment loads hoisted ahead of the WMMA bursts.
__global__ void __launch_bounds__(256)
k_attn(const __bf16* __restrict__ qb, const __bf16* __restrict__ kbb,
       const __bf16* __restrict__ vbt, __bf16* __restrict__ ab) {
  __shared__ __bf16 plds[8][16 * 32];
  // per wave, per buffer: K tile (32x64 = 2048) then V tile (64x32 = 2048)
  __shared__ __bf16 kvs[8][2][4096];

  const int lane = threadIdx.x & 31;
  const int wsl  = threadIdx.x >> 5;
  const int wid  = blockIdx.x * 8 + wsl;
  const int qt = wid & 127;          // 128 query tiles per head
  const int h  = (wid >> 7) & 15;
  const int b  = wid >> 11;
  const int g  = h >> 2;             // kv head
  const int half = lane >> 4;
  const int lrow = lane & 15;
  const int kb   = half ? 8 : 0;

  // Q fragments: 16x64 = two K=32 A-fragments (reused across whole loop)
  v16bf qa[2];
  {
    const __bf16* qbase = qb + ((size_t)(b * T + qt * 16 + lrow)) * CD + h * HD;
    qa[0] = frag_from(qbase + kb);
    qa[1] = frag_from(qbase + 32 + kb);
  }

  const __bf16* krow = kbb + (size_t)b * T * KVD + g * HD;            // key-major
  const __bf16* vrow = vbt + ((size_t)(b * KV + g) * HD) * T;         // d-major

  v8f o0 = {}, o1 = {}, o2 = {}, o3 = {};
  float lsum[8], mrow[8];
#pragma unroll
  for (int mp = 0; mp < 8; ++mp) { lsum[mp] = 0.f; mrow[mp] = -3.0e38f; }

  const float scale = 0.125f; // 1/sqrt(64)
  const int kend = (qt + 1) * 16;

  // prologue: stage keys/values [0,32) into buffer 0
  tdm_load_2d(lds_off(&kvs[wsl][0][0]),    krow, 64, 32, (u32)KVD);   // K: 32 keys x 64 d
  tdm_load_2d(lds_off(&kvs[wsl][0][2048]), vrow, 32, 64, (u32)T);     // V: 64 d x 32 keys

  int buf = 0;
  for (int j0 = 0; j0 < kend; j0 += 32) {
    if (j0 + 32 < kend) {
      tdm_load_2d(lds_off(&kvs[wsl][buf ^ 1][0]),    krow + (size_t)(j0 + 32) * KVD, 64, 32, (u32)KVD);
      tdm_load_2d(lds_off(&kvs[wsl][buf ^ 1][2048]), vrow + (j0 + 32),               32, 64, (u32)T);
      __builtin_amdgcn_s_wait_tensorcnt(2);
    } else {
      __builtin_amdgcn_s_wait_tensorcnt(0);
    }
    asm volatile("" ::: "memory");

    const __bf16* Ks  = &kvs[wsl][buf][0];     // [key][d]  32x64
    const __bf16* VsT = Ks + 2048;             // [d][key]  64x32

    // ---- S = Q @ K^T for 32 keys: hoist 4 K fragments, then 4 WMMAs ----
    const v16bf kf0 = frag_from(Ks + lrow * 64 + kb);               // keys 0-15, d 0-31
    const v16bf kf1 = frag_from(Ks + (16 + lrow) * 64 + kb);        // keys 16-31, d 0-31
    const v16bf kf2 = frag_from(Ks + lrow * 64 + 32 + kb);          // keys 0-15, d 32-63
    const v16bf kf3 = frag_from(Ks + (16 + lrow) * 64 + 32 + kb);   // keys 16-31, d 32-63
    v8f s0 = {}, s1 = {};
    s0 = __builtin_amdgcn_wmma_f32_16x16x32_bf16(false, qa[0], false, kf0, (short)0, s0, false, false);
    s1 = __builtin_amdgcn_wmma_f32_16x16x32_bf16(false, qa[0], false, kf1, (short)0, s1, false, false);
    s0 = __builtin_amdgcn_wmma_f32_16x16x32_bf16(false, qa[1], false, kf2, (short)0, s0, false, false);
    s1 = __builtin_amdgcn_wmma_f32_16x16x32_bf16(false, qa[1], false, kf3, (short)0, s1, false, false);

    // ---- online softmax (rows in VGPR index, cols across lanes) ----
#pragma unroll
    for (int mp = 0; mp < 8; ++mp) {
      const int qi  = qt * 16 + mp + 8 * half;
      const int k0i = j0 + lrow;
      const int k1i = j0 + 16 + lrow;
      float v0 = (k0i <= qi) ? s0[mp] * scale : -3.0e38f;
      float v1 = (k1i <= qi) ? s1[mp] * scale : -3.0e38f;
      float cm = fmaxf(v0, v1);
#pragma unroll
      for (int off = 1; off < 16; off <<= 1)
        cm = fmaxf(cm, __shfl_xor(cm, off, 32));
      const float mn    = fmaxf(mrow[mp], cm);
      const float alpha = __expf(mrow[mp] - mn);
      mrow[mp] = mn;
      const float p0 = __expf(v0 - mn);
      const float p1 = __expf(v1 - mn);
      float rs = p0 + p1;
#pragma unroll
      for (int off = 1; off < 16; off <<= 1)
        rs += __shfl_xor(rs, off, 32);
      lsum[mp] = lsum[mp] * alpha + rs;
      o0[mp] *= alpha; o1[mp] *= alpha; o2[mp] *= alpha; o3[mp] *= alpha;
      const int m = mp + 8 * half;
      plds[wsl][m * 32 + lrow]      = (__bf16)p0;
      plds[wsl][m * 32 + 16 + lrow] = (__bf16)p1;
    }
    asm volatile("s_wait_dscnt 0" ::: "memory");

    // ---- P: C-layout -> A-fragment (16x32) via LDS ----
    const v16bf pa = frag_from(&plds[wsl][lrow * 32 + kb]);

    // ---- O += P @ V: hoist 4 V fragments, then 4 WMMAs ----
    v16bf vfr[4];
#pragma unroll
    for (int t = 0; t < 4; ++t)
      vfr[t] = frag_from(VsT + (t * 16 + lrow) * 32 + kb);
    o0 = __builtin_amdgcn_wmma_f32_16x16x32_bf16(false, pa, false, vfr[0], (short)0, o0, false, false);
    o1 = __builtin_amdgcn_wmma_f32_16x16x32_bf16(false, pa, false, vfr[1], (short)0, o1, false, false);
    o2 = __builtin_amdgcn_wmma_f32_16x16x32_bf16(false, pa, false, vfr[2], (short)0, o2, false, false);
    o3 = __builtin_amdgcn_wmma_f32_16x16x32_bf16(false, pa, false, vfr[3], (short)0, o3, false, false);
    buf ^= 1;
  }

  // ---- normalize + store attention output (bf16, (B,T,C)) ----
#pragma unroll
  for (int mp = 0; mp < 8; ++mp) {
    const float inv = 1.0f / lsum[mp];
    const int tq = qt * 16 + mp + 8 * half;
    const size_t rowoff = ((size_t)(b * T + tq)) * CD + h * HD;
    ab[rowoff + 0 * 16 + lrow] = (__bf16)(o0[mp] * inv);
    ab[rowoff + 1 * 16 + lrow] = (__bf16)(o1[mp] * inv);
    ab[rowoff + 2 * 16 + lrow] = (__bf16)(o2[mp] * inv);
    ab[rowoff + 3 * 16 + lrow] = (__bf16)(o3[mp] * inv);
  }
}

// ======================================================================
extern "C" void kernel_launch(void* const* d_in, const int* in_sizes, int n_in,
                              void* d_out, int out_size, void* d_ws, size_t ws_size,
                              hipStream_t stream) {
  const float* x  = (const float*)d_in[0];
  const float* wq = (const float*)d_in[1];
  const float* wk = (const float*)d_in[2];
  const float* wv = (const float*)d_in[3];
  const float* wo = (const float*)d_in[4];

  float* out  = (float*)d_out;                           // (B,T,C)
  float* kout = out + (size_t)MR * CD;                   // (B,KV,T,hd)
  float* vout = kout + (size_t)MR * KVD;                 // (B,KV,T,hd)

  // workspace carve-up
  char* w = (char*)d_ws;
  __bf16* xb   = (__bf16*)w;  w += (size_t)MR * CD * 2;
  __bf16* wqT  = (__bf16*)w;  w += (size_t)CD * CD * 2;     // (N=C) x K
  __bf16* wkT  = (__bf16*)w;  w += (size_t)CD * KVD * 2;    // (N=kv) x K
  __bf16* wvT  = (__bf16*)w;  w += (size_t)CD * KVD * 2;
  __bf16* woT  = (__bf16*)w;  w += (size_t)CD * CD * 2;
  float*  qf   = (float*)w;   w += (size_t)MR * CD * 4;
  float*  kf   = (float*)w;   w += (size_t)MR * KVD * 4;
  float*  vf   = (float*)w;   w += (size_t)MR * KVD * 4;
  __bf16* qbb  = (__bf16*)w;  w += (size_t)MR * CD * 2;
  __bf16* kbb  = (__bf16*)w;  w += (size_t)MR * KVD * 2;
  __bf16* vbt  = (__bf16*)w;  w += (size_t)MR * KVD * 2;    // d-major V
  __bf16* ab   = (__bf16*)w;  w += (size_t)MR * CD * 2;

  const int TPB = 256;
  auto blocksFor = [](int n) { return (n + 255) / 256; };

  // 1) bf16 conversion: x row-major, weights transposed to N x K
  k_tobf16<<<blocksFor(MR * CD),   TPB, 0, stream>>>(x, xb, MR * CD);
  k_tobf16T<<<blocksFor(CD * CD),  TPB, 0, stream>>>(wq, wqT, CD, CD);
  k_tobf16T<<<blocksFor(CD * KVD), TPB, 0, stream>>>(wk, wkT, CD, KVD);
  k_tobf16T<<<blocksFor(CD * KVD), TPB, 0, stream>>>(wv, wvT, CD, KVD);
  k_tobf16T<<<blocksFor(CD * CD),  TPB, 0, stream>>>(wo, woT, CD, CD);

  // 2) projections: waves = (M/32)*(N/64), 8 waves per block
  k_gemm<<<(MR / 32) * (CD  / 64) / 8, TPB, 0, stream>>>(xb, wqT, qf, CD,  CD);
  k_gemm<<<(MR / 32) * (KVD / 64) / 8, TPB, 0, stream>>>(xb, wkT, kf, KVD, CD);
  k_gemm<<<(MR / 32) * (KVD / 64) / 8, TPB, 0, stream>>>(xb, wvT, vf, KVD, CD);

  // 3) RoPE + cache writes
  k_rope_q<<<blocksFor(MR * CD),  TPB, 0, stream>>>(qf, qbb);
  k_rope_k<<<blocksFor(MR * KVD), TPB, 0, stream>>>(kf, kbb, kout);
  k_vcache<<<blocksFor(MR * KVD), TPB, 0, stream>>>(vf, vbt, vout);

  // 4) flash attention: B*H*(T/16) = 4096 waves, 8 per block
  k_attn<<<(Bn * H * (T / 16)) / 8, TPB, 0, stream>>>(qbb, kbb, vbt, ab);

  // 5) output projection
  k_gemm<<<(MR / 32) * (CD / 64) / 8, TPB, 0, stream>>>(ab, woT, out, CD, CD);
}
